// RelGraphConvLayer_14783277433376
// MI455X (gfx1250) — matrix operand order repaired and checked by reading, compiled, not run on hardware
//
#include <hip/hip_runtime.h>
#include <hip/hip_bf16.h>

typedef __attribute__((ext_vector_type(16))) __bf16 v16bf;
typedef __attribute__((ext_vector_type(8)))  __bf16 v8bf;
typedef __attribute__((ext_vector_type(8)))  float  v8f;

#define N_NODES_C 50000
#define D_C       128
#define N_EDGES_C 800000
#define EPW       128   // edges per wave in scatter phase

// ---------------- Phase 0: zero workspace ----------------
__global__ void zero_ws_kernel(float* __restrict__ p, long n) {
    long i = (long)blockIdx.x * blockDim.x + threadIdx.x;
    long stride = (long)gridDim.x * blockDim.x;
    for (; i < n; i += stride) p[i] = 0.0f;
}

// ---------------- Phase 0b: pre-pack weights into bf16 WMMA B-fragment order ----
// Layout: Wpk[mat][t][ntile][lane][i], i=0..15 contiguous per lane (32 B).
// Fragment semantics: K = 32*t + 8*(lane>>4) + (i&7) + 16*(i>>3), n = 16*ntile + (lane&15).
__global__ void pack_w_kernel(const float* __restrict__ W0,
                              const float* __restrict__ W1,
                              const float* __restrict__ W2,
                              const float* __restrict__ Wl,
                              __bf16*      __restrict__ Wpk) {
    int idx = blockIdx.x * blockDim.x + threadIdx.x;   // 4*4*8*32*16 = 65536
    int i    = idx & 15;
    int lane = (idx >> 4) & 31;
    int w    = (idx >> 9) & 7;
    int t    = (idx >> 12) & 3;
    int mat  = idx >> 14;
    const float* Ws[4] = {W0, W1, W2, Wl};
    int K = 32 * t + 8 * (lane >> 4) + (i & 7) + 16 * (i >> 3);
    int n = 16 * w + (lane & 15);
    Wpk[idx] = (__bf16)Ws[mat][K * D_C + n];
}

// ---------------- Phase 1: segmented scatter-add (dst is sorted) ----------------
// One wave handles EPW contiguous edges. Each lane owns 4 features (float4).
// Runs of equal dst are accumulated in registers; flush = 4 global f32 atomics
// per lane + 1 count atomic per run.
__global__ void scatter_kernel(const float* __restrict__ x,
                               const int*   __restrict__ src,
                               const int*   __restrict__ dst,
                               float*       __restrict__ agg,
                               float*       __restrict__ cnt,
                               int n_edges) {
    const int lane = threadIdx.x & 31;
    const int wave = blockIdx.x * (blockDim.x >> 5) + (threadIdx.x >> 5);
    int e0 = wave * EPW;
    if (e0 >= n_edges) return;
    int e1 = e0 + EPW; if (e1 > n_edges) e1 = n_edges;

    const float4* x4 = (const float4*)x;

    float4 acc = make_float4(0.f, 0.f, 0.f, 0.f);
    int cur = dst[e0];   // wave-uniform
    int run = 0;

    for (int e = e0; e < e1; ++e) {
        int d = dst[e];
        if (d != cur) {
            float* a = agg + (long)cur * D_C + lane * 4;
            atomicAdd(a + 0, acc.x);
            atomicAdd(a + 1, acc.y);
            atomicAdd(a + 2, acc.z);
            atomicAdd(a + 3, acc.w);
            if (lane == 0) atomicAdd(cnt + cur, (float)run);
            acc = make_float4(0.f, 0.f, 0.f, 0.f);
            run = 0;
            cur = d;
        }
        int s = src[e];  // wave-uniform
        float4 v = x4[(long)s * 32 + lane];
        acc.x += v.x; acc.y += v.y; acc.z += v.z; acc.w += v.w;
        ++run;
    }
    float* a = agg + (long)cur * D_C + lane * 4;
    atomicAdd(a + 0, acc.x);
    atomicAdd(a + 1, acc.y);
    atomicAdd(a + 2, acc.z);
    atomicAdd(a + 3, acc.w);
    if (lane == 0) atomicAdd(cnt + cur, (float)run);
}

// ---------------- Phase 2: fused 4-way GEMM + bias + ReLU via WMMA ----------------
// Block = 256 threads = 8 waves. Block owns 16 output rows; wave w owns
// columns [16w, 16w+16). Accumulate over mats {mean0,mean1,mean2,x} with
// weights {W0,W1,W2,Wl}: 4 mats x 4 K-steps of v_wmma_f32_16x16x32_bf16.
__global__ void gemm_kernel(const float*  __restrict__ x,
                            const float*  __restrict__ agg,   // 3 contiguous N x D matrices
                            const float*  __restrict__ cnt,   // 3 contiguous N vectors
                            const __bf16* __restrict__ Wpk,   // packed bf16 B fragments
                            const float*  __restrict__ bl,
                            float*        __restrict__ out) {
    __shared__ __align__(16) __bf16 A_lds[4][16][D_C];   // 16 KB

    const int tid  = threadIdx.x;
    const int row0 = blockIdx.x * 16;

    // Stage the four 16x128 A tiles into LDS as bf16 (means scaled by 1/max(cnt,1)).
    for (int idx = tid; idx < 4 * 16 * D_C; idx += blockDim.x) {
        int mat = idx >> 11;          // /2048
        int rem = idx & 2047;
        int r   = rem >> 7;           // /128
        int c   = rem & 127;
        int row = row0 + r;
        float v;
        if (mat < 3) {
            float cc  = cnt[mat * N_NODES_C + row];
            float inv = 1.0f / fmaxf(cc, 1.0f);
            v = agg[((long)mat * N_NODES_C + row) * D_C + c] * inv;
        } else {
            v = x[(long)row * D_C + c];
        }
        A_lds[mat][r][c] = (__bf16)v;
    }
    __syncthreads();

    const int wave = tid >> 5;
    const int lane = tid & 31;
    const int half = lane >> 4;       // 0: lanes 0-15, 1: lanes 16-31
    const int nloc = lane & 15;
    const int n    = wave * 16 + nloc;

    v8f c = {};

#pragma unroll
    for (int mat = 0; mat < 4; ++mat) {
#pragma unroll
        for (int t = 0; t < 4; ++t) {
            // A fragment (ISA 16-bit A 16x32 layout): lane holds row m = lane%16,
            // elems 0..7 = K 32t + 8*half + i; elems 8..15 = K 32t + 16 + 8*half + i.
            int k0 = 32 * t + 8 * half;
            v8bf lo = *(const v8bf*)&A_lds[mat][nloc][k0];
            v8bf hi = *(const v8bf*)&A_lds[mat][nloc][k0 + 16];
            v16bf a = __builtin_shufflevector(lo, hi,
                0, 1, 2, 3, 4, 5, 6, 7, 8, 9, 10, 11, 12, 13, 14, 15);

            // B fragment: one contiguous 32 B per-lane load from the packed buffer.
            v16bf b = *(const v16bf*)(Wpk +
                ((size_t)((((mat * 4 + t) * 8 + wave) * 32) + lane) << 4));

            c = __builtin_amdgcn_wmma_f32_16x16x32_bf16(
                    false, a, false, b, (short)0, c, false, false);
        }
    }

    // C/D layout: VGPR r -> row (r + 8*half), col n. Fuse bias + ReLU.
    float bias = bl[n];
#pragma unroll
    for (int r = 0; r < 8; ++r) {
        int row = row0 + r + 8 * half;
        float v = c[r] + bias;
        out[(long)row * D_C + n] = fmaxf(v, 0.0f);
    }
}

extern "C" void kernel_launch(void* const* d_in, const int* in_sizes, int n_in,
                              void* d_out, int out_size, void* d_ws, size_t ws_size,
                              hipStream_t stream) {
    const float* x    = (const float*)d_in[0];
    const int*   src0 = (const int*)  d_in[1];
    const int*   dst0 = (const int*)  d_in[2];
    const int*   src1 = (const int*)  d_in[3];
    const int*   dst1 = (const int*)  d_in[4];
    const int*   src2 = (const int*)  d_in[5];
    const int*   dst2 = (const int*)  d_in[6];
    const float* W0   = (const float*)d_in[7];
    const float* W1   = (const float*)d_in[8];
    const float* W2   = (const float*)d_in[9];
    const float* Wl   = (const float*)d_in[10];
    const float* bl   = (const float*)d_in[11];
    float* out = (float*)d_out;

    // Workspace layout:
    //   [agg0|agg1|agg2]  3 * N * D f32   (76.8 MB)
    //   [cnt0|cnt1|cnt2]  3 * N     f32   (600 KB)
    //   [Wpk]             4*4*8*32*16 bf16 (128 KB), 32B-aligned offset
    float*  agg = (float*)d_ws;
    float*  cnt = agg + (size_t)3 * N_NODES_C * D_C;
    __bf16* Wpk = (__bf16*)(cnt + (size_t)3 * N_NODES_C);

    long zn = (long)3 * N_NODES_C * D_C + 3 * N_NODES_C;
    zero_ws_kernel<<<2048, 256, 0, stream>>>(agg, zn);

    pack_w_kernel<<<65536 / 256, 256, 0, stream>>>(W0, W1, W2, Wl, Wpk);

    int waves  = (N_EDGES_C + EPW - 1) / EPW;   // 6250 waves per relation
    int blocks = (waves + 7) / 8;               // 8 waves per 256-thread block
    scatter_kernel<<<blocks, 256, 0, stream>>>(x, src0, dst0,
        agg,                               cnt,                 N_EDGES_C);
    scatter_kernel<<<blocks, 256, 0, stream>>>(x, src1, dst1,
        agg + (size_t)1 * N_NODES_C * D_C, cnt + 1 * N_NODES_C, N_EDGES_C);
    scatter_kernel<<<blocks, 256, 0, stream>>>(x, src2, dst2,
        agg + (size_t)2 * N_NODES_C * D_C, cnt + 2 * N_NODES_C, N_EDGES_C);

    gemm_kernel<<<N_NODES_C / 16, 256, 0, stream>>>(x, agg, cnt,
        Wpk, bl, out);
}
